// TransformerCrossAttnLayer_78091095376089
// MI455X (gfx1250) — matrix operand order repaired and checked by reading, not compiled, not run on hardware
//
#include <hip/hip_runtime.h>
#include <hip/hip_bf16.h>
#include <math.h>

#define NTOT  80      // bs*h
#define WDIM  256     // w
#define NHEAD 8
#define DIM   16
#define CDIM  128
#define FFDIM 512

typedef __attribute__((ext_vector_type(2)))  float  v2f;
typedef __attribute__((ext_vector_type(4)))  float  v4f;
typedef __attribute__((ext_vector_type(8)))  float  v8f;
typedef __attribute__((ext_vector_type(4)))  __bf16 v4bf;
typedef __attribute__((ext_vector_type(8)))  __bf16 v8bf;
typedef __attribute__((ext_vector_type(16))) __bf16 v16bf;

union BF16x16 { v16bf v; v8bf h[2]; };

__device__ __forceinline__ float wave_sum(float v) {
#pragma unroll
  for (int o = 16; o > 0; o >>= 1) v += __shfl_xor(v, o, 32);
  return v;
}
__device__ __forceinline__ float wave_max(float v) {
#pragma unroll
  for (int o = 16; o > 0; o >>= 1) v = fmaxf(v, __shfl_xor(v, o, 32));
  return v;
}

// ---------------------------------------------------------------- kernel 1
// Convert FF weights to bf16 AND transpose to [col][k] so WMMA B-fragments
// are two contiguous 8-element runs of K per lane.
__global__ __launch_bounds__(256)
void cvt_weights(const float* __restrict__ w1, const float* __restrict__ w2,
                 __bf16* __restrict__ w1t, __bf16* __restrict__ w2t) {
  int idx = blockIdx.x * 256 + threadIdx.x;          // 0 .. 65535
  if (idx < CDIM * FFDIM) {
    int c1 = idx / CDIM, k1 = idx % CDIM;            // w1t[512][128]
    w1t[idx] = (__bf16)w1[k1 * FFDIM + c1];
    int c2 = idx / FFDIM, k2 = idx % FFDIM;          // w2t[128][512]
    w2t[idx] = (__bf16)w2[k2 * CDIM + c2];
  }
}

// ---------------------------------------------------------------- kernel 2
// Per-head KQV projection: [16 rows x 16 dim] @ [16 x 48] in fp32 WMMA.
// Q,K out: [n,h,w,d] f32.  V out: [n,h,d,w] bf16 (pre-transposed for AV).
__global__ __launch_bounds__(32)
void kqv_kernel(const float* __restrict__ fL, const float* __restrict__ fR,
                const float* __restrict__ kqvw,
                float* __restrict__ qL, float* __restrict__ qR,
                float* __restrict__ kL, float* __restrict__ kR,
                __bf16* __restrict__ vL, __bf16* __restrict__ vR) {
  __shared__ __align__(16) float As[16][DIM];
  __shared__ __align__(16) float Bst[48][DIM];       // transposed weight [col][k]

  const int lane = threadIdx.x;
  const int head = blockIdx.y;
  const int side = blockIdx.z;
  const float* feat = side ? fR : fL;
  float*  qb = side ? qR : qL;
  float*  kb = side ? kR : kL;
  __bf16* vb = side ? vR : vL;
  const int r0  = blockIdx.x * 16;
  const int n   = r0 / WDIM;                         // whole tile in one n
  const int wc0 = r0 % WDIM;

#pragma unroll
  for (int i = 0; i < 2; ++i) {                      // A tile, b128 loads
    int e4 = i * 32 + lane;                          // 0..63 float4 slots
    int r = e4 >> 2, d0 = (e4 & 3) * 4;
    *(v4f*)&As[r][d0] =
        *(const v4f*)&feat[(long)(r0 + r) * CDIM + head * DIM + d0];
  }
#pragma unroll
  for (int i = 0; i < 24; ++i) {                     // weight, transposed stage
    int e = i * 32 + lane;                           // 0..767
    int c = e >> 4, k = e & 15;
    Bst[c][k] = kqvw[k * 48 + c];
  }
  __syncthreads();

  const int Mrow = lane & 15;
  const int hiHalf = lane >> 4;

  // lane-invariant output bases (offsets per vv are small constants)
  float* qkBase0 = (float*)0;                        // set per j below
  __bf16* vBase = vb + ((long)(n * NHEAD + head) * DIM + Mrow) * WDIM
                     + wc0 + hiHalf * 8;             // + vv
  const long qkOff = ((long)(n * NHEAD + head) * WDIM + wc0 + hiHalf * 8) * DIM
                     + Mrow;                          // + vv*DIM

#pragma unroll
  for (int j = 0; j < 3; ++j) {                      // 0=key 1=query 2=value
    v8f acc = {};
#pragma unroll
    for (int kc = 0; kc < 4; ++kc) {                 // K=16 in 4 chunks of 4
      int k0 = kc * 4 + hiHalf * 2;
      v2f a = *(const v2f*)&As[Mrow][k0];
      v2f b = *(const v2f*)&Bst[j * 16 + Mrow][k0];
      acc = __builtin_amdgcn_wmma_f32_16x16x4_f32(false, a, false, b,
                                                  (short)0, acc, false, false);
    }
    if (j == 2) {
#pragma unroll
      for (int vv = 0; vv < 8; ++vv) vBase[vv] = (__bf16)acc[vv];
    } else {
      qkBase0 = (j == 0 ? kb : qb) + qkOff;
#pragma unroll
      for (int vv = 0; vv < 8; ++vv) qkBase0[vv * DIM] = acc[vv];
    }
  }
}

// ---------------------------------------------------------------- kernel 3
// One wave: 16 query rows x full s=256. fp32 WMMA scores (+prev -> raw),
// wave softmax, bf16 WMMA for A*V.
__global__ __launch_bounds__(32)
void attn_kernel(const float* __restrict__ qL, const float* __restrict__ qR,
                 const float* __restrict__ kL, const float* __restrict__ kR,
                 const __bf16* __restrict__ vL, const __bf16* __restrict__ vR,
                 const float* __restrict__ prevL, const float* __restrict__ prevR,
                 float* __restrict__ rawL, float* __restrict__ rawR,
                 float* __restrict__ msgL, float* __restrict__ msgR) {
  __shared__ __align__(16) float  Qs[16][DIM];
  __shared__ __align__(16) float  Ks[WDIM][DIM];
  __shared__ __align__(16) __bf16 Vt[DIM][WDIM];     // transposed V
  __shared__ __align__(16) float  Ss[16][WDIM];
  __shared__ __align__(16) __bf16 Ps[16][WDIM];

  const int lane = threadIdx.x;
  const int lt   = blockIdx.x;
  const int nh   = blockIdx.y;
  const int n    = nh / NHEAD;
  const int head = nh % NHEAD;
  const int side = blockIdx.z;

  const float*  q    = side ? qR    : qL;
  const float*  k    = side ? kL    : kR;            // cross: opposite side
  const __bf16* v    = side ? vL    : vR;
  const float*  prev = side ? prevR : prevL;
  float*        raw  = side ? rawR  : rawL;
  float*        msg  = side ? msgR  : msgL;

  const int  l0     = lt * 16;
  const long basehw = (long)(n * NHEAD + head) * WDIM;

#pragma unroll
  for (int i = 0; i < 2; ++i) {                      // Q 16x16, b128 loads
    int e4 = i * 32 + lane;
    int r = e4 >> 2, d0 = (e4 & 3) * 4;
    *(v4f*)&Qs[r][d0] = *(const v4f*)&q[(basehw + l0 + r) * DIM + d0];
  }
#pragma unroll
  for (int i = 0; i < 32; ++i) {                     // K 256x16, b128 loads
    int e4 = i * 32 + lane;                          // 0..1023
    int s = e4 >> 2, d0 = (e4 & 3) * 4;
    *(v4f*)&Ks[s][d0] = *(const v4f*)&k[(basehw + s) * DIM + d0];
  }
#pragma unroll
  for (int i = 0; i < 16; ++i) {                     // Vt 16x256 (global layout
    int e8 = i * 32 + lane;                          // [n,h,d,w] -> coalesced)
    int d = e8 >> 5, s0 = (e8 & 31) * 8;
    *(v8bf*)&Vt[d][s0] = *(const v8bf*)&v[(basehw + d) * 0 +   // keep simple:
        ((long)(n * NHEAD + head) * DIM + d) * WDIM + s0];
  }
  __syncthreads();

  const int Mrow = lane & 15;
  const int hiHalf = lane >> 4;

  // lane-invariant prev/raw base: element (m=vv+hiHalf*8, s=st*16+Mrow)
  //   index = base + vv*(WDIM*NHEAD) + st*(16*NHEAD)
  const long pBase = (((long)n * WDIM + l0 + hiHalf * 8) * WDIM + Mrow) * NHEAD
                     + head;
  const float* prevB = prev + pBase;
  float*       rawB  = raw  + pBase;

  // ---- scores: 16 s-tiles, K=16 via 4x fp32 WMMA each
  for (int st = 0; st < 16; ++st) {
    const float* pp = prevB + st * (16 * NHEAD);
    float*       rp = rawB  + st * (16 * NHEAD);
    if (st < 15) __builtin_prefetch(pp + 16 * NHEAD, 0, 1);
    v8f acc = {};
#pragma unroll
    for (int kc = 0; kc < 4; ++kc) {
      int k0 = kc * 4 + hiHalf * 2;
      v2f a = *(const v2f*)&Qs[Mrow][k0];
      v2f b = *(const v2f*)&Ks[st * 16 + Mrow][k0];
      acc = __builtin_amdgcn_wmma_f32_16x16x4_f32(false, a, false, b,
                                                  (short)0, acc, false, false);
    }
#pragma unroll
    for (int vv = 0; vv < 8; ++vv) {
      float r = acc[vv] + pp[vv * (WDIM * NHEAD)];
      rp[vv * (WDIM * NHEAD)] = r;                   // returned raw scores
      Ss[vv + hiHalf * 8][st * 16 + Mrow] = r;
    }
  }
  __syncthreads();

  // ---- softmax(0.25 * raw) over s, per row, wave-parallel
  for (int m = 0; m < 16; ++m) {
    float t[8], mx = -INFINITY;
#pragma unroll
    for (int i = 0; i < 8; ++i) {
      t[i] = 0.25f * Ss[m][lane + 32 * i];
      mx = fmaxf(mx, t[i]);
    }
    mx = wave_max(mx);
    float sum = 0.f;
#pragma unroll
    for (int i = 0; i < 8; ++i) { t[i] = __expf(t[i] - mx); sum += t[i]; }
    sum = wave_sum(sum);
    float inv = 1.0f / sum;
#pragma unroll
    for (int i = 0; i < 8; ++i) Ps[m][lane + 32 * i] = (__bf16)(t[i] * inv);
  }
  __syncthreads();

  // ---- msg = P @ V : 8 chunks of K=32, bf16 WMMA, all-b128 fragments
  v8f acc = {};
  for (int sc = 0; sc < 8; ++sc) {
    BF16x16 a, b;
    int ka0 = sc * 32 + hiHalf * 8;                  // A: two 8-elem runs
    a.h[0] = *(const v8bf*)&Ps[Mrow][ka0];
    a.h[1] = *(const v8bf*)&Ps[Mrow][ka0 + 16];
    int kb0 = sc * 32 + hiHalf * 8;                  // B: two 8-elem runs
    b.h[0] = *(const v8bf*)&Vt[Mrow][kb0];
    b.h[1] = *(const v8bf*)&Vt[Mrow][kb0 + 16];
    acc = __builtin_amdgcn_wmma_f32_16x16x32_bf16(false, a.v, false, b.v,
                                                  (short)0, acc, false, false);
  }
  float* msgB = msg + ((long)n * WDIM + l0 + hiHalf * 8) * CDIM
                    + head * DIM + Mrow;
#pragma unroll
  for (int vv = 0; vv < 8; ++vv) msgB[vv * CDIM] = acc[vv];
}

// ---------------------------------------------------------------- kernel 4
// LN1 -> GEMM(128->512)+GELU -> GEMM(512->128) -> residual -> LN2
__global__ __launch_bounds__(256)
void ffn_kernel(const float* __restrict__ fL, const float* __restrict__ fR,
                const float* __restrict__ msgL, const float* __restrict__ msgR,
                const __bf16* __restrict__ w1t, const __bf16* __restrict__ w2t,
                const float* __restrict__ b1, const float* __restrict__ b2,
                const float* __restrict__ g1, const float* __restrict__ be1,
                const float* __restrict__ g2, const float* __restrict__ be2,
                float* __restrict__ outL, float* __restrict__ outR) {
  __shared__ __align__(16) float  xs[16][CDIM];
  __shared__ __align__(16) __bf16 x1b[16][CDIM];
  __shared__ __align__(16) __bf16 hdnb[16][FFDIM];

  const int tid  = threadIdx.x;
  const int lane = tid & 31, wv = tid >> 5;
  const int side = blockIdx.z;
  const float* feat = side ? fR : fL;
  const float* msg  = side ? msgR : msgL;
  float*       out  = side ? outR : outL;
  const int r0 = blockIdx.x * 16;

#pragma unroll
  for (int i = 0; i < 2; ++i) {                      // x = feat + msg (b128)
    int e4 = i * 256 + tid;                          // 0..511 float4 slots
    int r = e4 >> 5, c0 = (e4 & 31) * 4;
    long gi = (long)(r0 + r) * CDIM + c0;
    *(v4f*)&xs[r][c0] = *(const v4f*)&feat[gi] + *(const v4f*)&msg[gi];
  }
  __syncthreads();

  // LN1 (wave wv -> rows 2wv, 2wv+1), each lane owns 4 contiguous channels
  for (int rr = 0; rr < 2; ++rr) {
    int row = wv * 2 + rr;
    v4f val = *(const v4f*)&xs[row][lane * 4];
    float s  = wave_sum(val.x + val.y + val.z + val.w);
    float s2 = wave_sum(val.x * val.x + val.y * val.y +
                        val.z * val.z + val.w * val.w);
    float mu   = s * (1.0f / CDIM);
    float var  = s2 * (1.0f / CDIM) - mu * mu;
    float rstd = rsqrtf(var + 1e-5f);
    v4f g = *(const v4f*)&g1[lane * 4];
    v4f bb = *(const v4f*)&be1[lane * 4];
    v4f xn = (val - mu) * rstd * g + bb;
    *(v4f*)&xs[row][lane * 4] = xn;                  // x1 kept for residual
    v4bf xb = { (__bf16)xn.x, (__bf16)xn.y, (__bf16)xn.z, (__bf16)xn.w };
    *(v4bf*)&x1b[row][lane * 4] = xb;
  }
  __syncthreads();

  const int Mrow = lane & 15, hiHalf = lane >> 4;

  // GEMM1: [16x128] @ [128x512], 32 col tiles over 8 waves
  for (int tc = 0; tc < 4; ++tc) {
    int ct = wv * 4 + tc;
    int col = ct * 16 + Mrow;
    const __bf16* wcol = w1t + (long)col * CDIM;
    v8f acc = {};
    for (int kc = 0; kc < 4; ++kc) {
      BF16x16 a, b;
      int ka0 = kc * 32 + hiHalf * 8;
      a.h[0] = *(const v8bf*)&x1b[Mrow][ka0];
      a.h[1] = *(const v8bf*)&x1b[Mrow][ka0 + 16];
      int kb0 = kc * 32 + hiHalf * 8;                // two 8-elem K runs
      b.h[0] = *(const v8bf*)&wcol[kb0];
      b.h[1] = *(const v8bf*)&wcol[kb0 + 16];
      acc = __builtin_amdgcn_wmma_f32_16x16x32_bf16(false, a.v, false, b.v,
                                                    (short)0, acc, false, false);
    }
    float bias = b1[col];
    __bf16* hb = &hdnb[hiHalf * 8][col];
#pragma unroll
    for (int vv = 0; vv < 8; ++vv) {
      float h = acc[vv] + bias;
      float gl = 0.5f * h * (1.0f + erff(h * 0.70710678f));   // exact gelu
      hb[vv * FFDIM] = (__bf16)gl;
    }
  }
  __syncthreads();

  // GEMM2: [16x512] @ [512x128], 8 col tiles, one per wave
  {
    int col = wv * 16 + Mrow;
    const __bf16* wcol = w2t + (long)col * FFDIM;
    v8f acc = {};
    for (int kc = 0; kc < 16; ++kc) {
      BF16x16 a, b;
      int ka0 = kc * 32 + hiHalf * 8;
      a.h[0] = *(const v8bf*)&hdnb[Mrow][ka0];
      a.h[1] = *(const v8bf*)&hdnb[Mrow][ka0 + 16];
      int kb0 = kc * 32 + hiHalf * 8;
      b.h[0] = *(const v8bf*)&wcol[kb0];
      b.h[1] = *(const v8bf*)&wcol[kb0 + 16];
      acc = __builtin_amdgcn_wmma_f32_16x16x32_bf16(false, a.v, false, b.v,
                                                    (short)0, acc, false, false);
    }
    float bias = b2[col];
    float* xp = &xs[hiHalf * 8][col];
#pragma unroll
    for (int vv = 0; vv < 8; ++vv)
      xp[vv * CDIM] = xp[vv * CDIM] + acc[vv] + bias;  // residual, same-lane RW
  }
  __syncthreads();

  // LN2 -> output
  for (int rr = 0; rr < 2; ++rr) {
    int row = wv * 2 + rr;
    v4f val = *(const v4f*)&xs[row][lane * 4];
    float s  = wave_sum(val.x + val.y + val.z + val.w);
    float s2 = wave_sum(val.x * val.x + val.y * val.y +
                        val.z * val.z + val.w * val.w);
    float mu   = s * (1.0f / CDIM);
    float var  = s2 * (1.0f / CDIM) - mu * mu;
    float rstd = rsqrtf(var + 1e-5f);
    v4f g = *(const v4f*)&g2[lane * 4];
    v4f bb = *(const v4f*)&be2[lane * 4];
    v4f xn = (val - mu) * rstd * g + bb;
    *(v4f*)&out[(long)(r0 + row) * CDIM + lane * 4] = xn;
  }
}

// ---------------------------------------------------------------- launch
extern "C" void kernel_launch(void* const* d_in, const int* in_sizes, int n_in,
                              void* d_out, int out_size, void* d_ws, size_t ws_size,
                              hipStream_t stream) {
  (void)in_sizes; (void)n_in; (void)out_size; (void)ws_size;
  const float* fL    = (const float*)d_in[0];
  const float* fR    = (const float*)d_in[1];
  const float* prevL = (const float*)d_in[2];
  const float* prevR = (const float*)d_in[3];
  const float* kqvw  = (const float*)d_in[4];
  const float* w1    = (const float*)d_in[5];
  const float* b1    = (const float*)d_in[6];
  const float* w2    = (const float*)d_in[7];
  const float* b2    = (const float*)d_in[8];
  const float* g1    = (const float*)d_in[9];
  const float* be1   = (const float*)d_in[10];
  const float* g2    = (const float*)d_in[11];
  const float* be2   = (const float*)d_in[12];

  const size_t QK_BYTES  = (size_t)NTOT * NHEAD * WDIM * DIM * sizeof(float);
  const size_t V_BYTES   = (size_t)NTOT * NHEAD * WDIM * DIM * sizeof(__bf16);
  const size_t MSG_BYTES = (size_t)NTOT * WDIM * CDIM * sizeof(float);
  const size_t W_BYTES   = (size_t)CDIM * FFDIM * sizeof(__bf16);

  char* ws = (char*)d_ws;
  size_t off = 0;
  auto carve = [&](size_t bytes) -> void* {
    void* p = ws + off;
    off += (bytes + 255) & ~(size_t)255;
    return p;
  };
  float*  qL   = (float*)carve(QK_BYTES);
  float*  qR   = (float*)carve(QK_BYTES);
  float*  kL   = (float*)carve(QK_BYTES);
  float*  kR   = (float*)carve(QK_BYTES);
  __bf16* vL   = (__bf16*)carve(V_BYTES);
  __bf16* vR   = (__bf16*)carve(V_BYTES);
  float*  msgL = (float*)carve(MSG_BYTES);
  float*  msgR = (float*)carve(MSG_BYTES);
  __bf16* w1t  = (__bf16*)carve(W_BYTES);
  __bf16* w2t  = (__bf16*)carve(W_BYTES);

  float* outAll = (float*)d_out;
  float* outL = outAll;
  float* outR = outAll + (size_t)NTOT * WDIM * CDIM;
  float* rawL = outR  + (size_t)NTOT * WDIM * CDIM;
  float* rawR = rawL  + (size_t)NTOT * WDIM * WDIM * NHEAD;

  cvt_weights<<<256, 256, 0, stream>>>(w1, w2, w1t, w2t);

  kqv_kernel<<<dim3(NTOT * WDIM / 16, NHEAD, 2), 32, 0, stream>>>(
      fL, fR, kqvw, qL, qR, kL, kR, vL, vR);

  attn_kernel<<<dim3(WDIM / 16, NTOT * NHEAD, 2), 32, 0, stream>>>(
      qL, qR, kL, kR, vL, vR, prevL, prevR, rawL, rawR, msgL, msgR);

  ffn_kernel<<<dim3(NTOT * WDIM / 16, 1, 2), 256, 0, stream>>>(
      fL, fR, msgL, msgR, w1t, w2t, b1, b2, g1, be1, g2, be2, outL, outR);
}
